// tauGRU_35046933135381
// MI455X (gfx1250) — matrix operand order repaired and checked by reading, not compiled, not run on hardware
//
#include <hip/hip_runtime.h>
#include <hip/hip_bf16.h>

// ---------------------------------------------------------------------------
// tauGRU scan on gfx1250 (MI455X).
// Persistent-kernel formulation: 32 workgroups x 256 threads (8 waves) stay
// resident for all 1024 timesteps, synchronizing once per step through an
// agent-scope atomic barrier. Weights are pre-packed to WMMA B-fragment
// layout and staged in LDS once (48 KB/WG), so the per-step critical path is
// h(global,f32) -> v_cvt -> v_wmma chain with B operands from ds_load_b128.
// 48 v_wmma_f32_16x16x32_f16 per wave per step. Next-step x rows are
// prefetched (global_prefetch_b8) during the current step's compute, and the
// epilogue's h_prev reads are hoisted above the GEMM so all global latency
// overlaps the WMMA chain.
// ---------------------------------------------------------------------------

typedef _Float16 f16;
typedef __attribute__((ext_vector_type(16))) _Float16 v16h;
typedef __attribute__((ext_vector_type(8)))  float    v8f;
typedef __attribute__((ext_vector_type(4)))  float    v4f;

#define T_STEPS 1024
#define BATCH   256
#define NINP    128
#define NHID    256
#define NOUT    10
#define KT_H    8              // 256/32 K-tiles over h
#define KT_ALL  12             // + 128/32 over x
#define CT      64             // (4*NHID)/16 gate-column tiles
#define NBLOCKS 32

// workspace layout (bytes, 256-aligned)
#define WP_HALFS   (KT_ALL * CT * 32 * 16)          // 393216 packed f16 weights
#define WP_BYTES   (WP_HALFS * 2)                   // 786432
#define BIAS_OFF   WP_BYTES                         // 1024 floats
#define HA_OFF     (BIAS_OFF + 4096)
#define H_BYTES    (BATCH * NHID * 4)               // 262144
#define HB_OFF     (HA_OFF + H_BYTES)
#define CNT_OFF    (HB_OFF + H_BYTES)               // barrier counter (uint)

__device__ __forceinline__ float sigm_f(float x) {
    return 1.0f / (1.0f + __expf(-x));
}
__device__ __forceinline__ float tanh_f(float x) {
    return 2.0f * sigm_f(2.0f * x) - 1.0f;   // tanh via fast exp
}

// Pack W_h [256,1024] and U_x [128,1024] (row-major fp32) into f16 B-fragment
// layout: Wp[kb][ct][lane][j], lane -> column (lane&15), K-in-tile = 8*(lane>>4)+j (j<8), 16+8*(lane>>4)+(j-8) (j>=8).
// Also fuse bias = b_h + b_x.
__global__ void tau_pack(const float* __restrict__ W_h, const float* __restrict__ U_x,
                         const float* __restrict__ b_h, const float* __restrict__ b_x,
                         f16* __restrict__ Wp, float* __restrict__ bias) {
    int idx = blockIdx.x * blockDim.x + threadIdx.x;
    if (idx < WP_HALFS) {
        int j    = idx & 15;
        int lane = (idx >> 4) & 31;
        int ct   = (idx >> 9) & 63;
        int kb   = idx >> 15;
        int c    = ct * 16 + (lane & 15);
        int kin  = (j < 8) ? (8 * (lane >> 4) + j) : (16 + 8 * (lane >> 4) + (j - 8));
        float v;
        if (kb < KT_H) {
            int k = kb * 32 + kin;
            v = W_h[k * 1024 + c];
        } else {
            int k = (kb - KT_H) * 32 + kin;
            v = U_x[k * 1024 + c];
        }
        Wp[idx] = (f16)v;
    }
    if (idx < 4 * NHID) bias[idx] = b_h[idx] + b_x[idx];
}

// Zero h0 and the grid-barrier counter (runs every launch: graph-replay safe).
__global__ void tau_init(float* __restrict__ h, unsigned* __restrict__ cnt) {
    int idx = blockIdx.x * blockDim.x + threadIdx.x;
    h[idx] = 0.0f;
    if (idx == 0) *cnt = 0u;
}

// Build 16x32 f16 A-fragment from a row-major fp32 row segment.
// Lanes hold row M = lane&15; halves 0..7 = K (8*half + 0..7),
// halves 8..15 = K (16 + 8*half + 0..7). p = row_base + k0 + 8*half.
__device__ __forceinline__ v16h load_a_frag(const float* __restrict__ p) {
    v4f f0 = *(const v4f*)(p);
    v4f f1 = *(const v4f*)(p + 4);
    v4f f2 = *(const v4f*)(p + 16);
    v4f f3 = *(const v4f*)(p + 20);
    v16h a;
    a[0]  = (f16)f0[0];  a[1]  = (f16)f0[1];  a[2]  = (f16)f0[2];  a[3]  = (f16)f0[3];
    a[4]  = (f16)f1[0];  a[5]  = (f16)f1[1];  a[6]  = (f16)f1[2];  a[7]  = (f16)f1[3];
    a[8]  = (f16)f2[0];  a[9]  = (f16)f2[1];  a[10] = (f16)f2[2];  a[11] = (f16)f2[3];
    a[12] = (f16)f3[0];  a[13] = (f16)f3[1];  a[14] = (f16)f3[2];  a[15] = (f16)f3[3];
    return a;
}

// Persistent scan kernel. Grid MUST be NBLOCKS x 256 (fully co-resident).
// wave wid = blockIdx*8 + wv; rt = wid&15 (batch tile), cth = wid>>4
// (= blockIdx>>1, shared by the whole block -> shared LDS B-fragments).
__global__ __launch_bounds__(256) void tau_scan(const float* __restrict__ input,
                                                const f16*   __restrict__ Wp,
                                                const float* __restrict__ bias,
                                                float* __restrict__ hA,
                                                float* __restrict__ hB,
                                                unsigned* __restrict__ cnt) {
    __shared__ f16 sB[KT_ALL * 4 * 32 * 16];   // 48 KB: B-fragments for this cth

    const int lane = threadIdx.x & 31;
    const int wv   = threadIdx.x >> 5;        // wave in block, 0..7
    const int cth  = blockIdx.x >> 1;         // hidden-column tile (0..15)
    const int rt   = ((blockIdx.x & 1) << 3) | wv;  // batch-row tile (0..15)
    const int n    = lane & 15;
    const int half = lane >> 4;
    const int m    = rt * 16 + n;             // this lane's A-matrix row

    // ---- stage this block's B-fragments into LDS (once) ----
    for (int f = wv; f < KT_ALL * 4; f += 8) {
        const int kb = f >> 2, g = f & 3;
        const f16* src = Wp + ((size_t)(kb * CT + g * 16 + cth) * 32 + lane) * 16;
        *(v16h*)(sB + ((size_t)f * 32 + lane) * 16) = *(const v16h*)src;
    }
    __syncthreads();

    const int col = cth * 16 + n;
    const float bA = bias[col];
    const float bB = bias[NHID + col];
    const float bC = bias[2 * NHID + col];
    const float bD = bias[3 * NHID + col];

    const float* hp = hA;
    float*       hn = hB;

    for (int t = 0; t < T_STEPS; ++t) {
        const float* xrow = input + (size_t)t * BATCH * NINP + m * NINP;
        const float* hrow = hp + m * NHID;

        // Prefetch next step's x rows (4 x 128B lines per row; two per lane)
        // so post-barrier x loads hit cache instead of HBM.
        {
            const int tn = (t + 1 < T_STEPS) ? (t + 1) : t;
            const float* xnext = input + (size_t)tn * BATCH * NINP + m * NINP;
            __builtin_prefetch(xnext + half * 64, 0, 0);
            __builtin_prefetch(xnext + half * 64 + 32, 0, 0);
        }

        // Hoist the epilogue's h_prev reads above the GEMM (strided by rows;
        // independent of the WMMA chain, so their latency is fully hidden).
        float hv[8];
#pragma unroll
        for (int r = 0; r < 8; ++r)
            hv[r] = hp[(rt * 16 + r + 8 * half) * NHID + col];

        v8f acc0 = {0.f,0.f,0.f,0.f,0.f,0.f,0.f,0.f};
        v8f acc1 = acc0, acc2 = acc0, acc3 = acc0;

#pragma unroll
        for (int kb = 0; kb < KT_ALL; ++kb) {
            const float* src = (kb < KT_H) ? (hrow + kb * 32)
                                           : (xrow + (kb - KT_H) * 32);
            v16h a = load_a_frag(src + half * 8);

            const f16* wb = sB + ((size_t)(kb * 4) * 32 + lane) * 16;
            v16h b0 = *(const v16h*)(wb);
            v16h b1 = *(const v16h*)(wb + 32 * 16);
            v16h b2 = *(const v16h*)(wb + 2 * 32 * 16);
            v16h b3 = *(const v16h*)(wb + 3 * 32 * 16);

            acc0 = __builtin_amdgcn_wmma_f32_16x16x32_f16(false, a, false, b0, (short)0, acc0, false, false);
            acc1 = __builtin_amdgcn_wmma_f32_16x16x32_f16(false, a, false, b1, (short)0, acc1, false, false);
            acc2 = __builtin_amdgcn_wmma_f32_16x16x32_f16(false, a, false, b2, (short)0, acc2, false, false);
            acc3 = __builtin_amdgcn_wmma_f32_16x16x32_f16(false, a, false, b3, (short)0, acc3, false, false);
        }

        // Epilogue. C/D layout: lane holds column n; VGPR r holds row
        // rt*16 + r + 8*half.
#pragma unroll
        for (int r = 0; r < 8; ++r) {
            const int row = rt * 16 + r + 8 * half;
            const float Av = tanh_f(acc0[r] + bA);
            const float Bv = tanh_f(acc1[r] + bB);
            const float Cv = sigm_f(acc2[r] + bC);   // DT = 1
            const float Dv = sigm_f(acc3[r] + bD);
            hn[row * NHID + col] = (1.0f - Cv) * hv[r] + Cv * (Av + Dv * Bv);
        }

        // ---- grid barrier: release stores, arrive, spin, acquire ----
        __builtin_amdgcn_fence(__ATOMIC_RELEASE, "agent");
        __syncthreads();
        if (threadIdx.x == 0) {
            const unsigned target = (unsigned)NBLOCKS * (unsigned)(t + 1);
            __hip_atomic_fetch_add(cnt, 1u, __ATOMIC_RELEASE, __HIP_MEMORY_SCOPE_AGENT);
            while (__hip_atomic_load(cnt, __ATOMIC_ACQUIRE, __HIP_MEMORY_SCOPE_AGENT) < target) {
                __builtin_amdgcn_s_sleep(2);
            }
        }
        __syncthreads();
        __builtin_amdgcn_fence(__ATOMIC_ACQUIRE, "agent");

        float* tmp = (float*)hp; hp = hn; hn = tmp;
    }
}

// out[b][o] = h[b]@W_out[:,o] + b_out[o]   (tiny, fp32 VALU is fine)
__global__ void tau_classifier(const float* __restrict__ h,
                               const float* __restrict__ W_out,
                               const float* __restrict__ b_out,
                               float* __restrict__ out) {
    const int b = blockIdx.x;
    const int o = threadIdx.x;
    if (o < NOUT) {
        float s = b_out[o];
        for (int k = 0; k < NHID; ++k)
            s += h[b * NHID + k] * W_out[k * NOUT + o];
        out[b * NOUT + o] = s;
    }
}

extern "C" void kernel_launch(void* const* d_in, const int* in_sizes, int n_in,
                              void* d_out, int out_size, void* d_ws, size_t ws_size,
                              hipStream_t stream) {
    (void)in_sizes; (void)n_in; (void)out_size; (void)ws_size;
    const float* input = (const float*)d_in[0];  // [T,B,128]
    const float* W_h   = (const float*)d_in[1];  // [256,1024]
    const float* b_h   = (const float*)d_in[2];  // [1024]
    const float* U_x   = (const float*)d_in[3];  // [128,1024]
    const float* b_x   = (const float*)d_in[4];  // [1024]
    const float* W_out = (const float*)d_in[5];  // [256,10]
    const float* b_out = (const float*)d_in[6];  // [10]
    float* out = (float*)d_out;

    char*     ws   = (char*)d_ws;
    f16*      Wp   = (f16*)ws;
    float*    bias = (float*)(ws + BIAS_OFF);
    float*    hA   = (float*)(ws + HA_OFF);
    float*    hB   = (float*)(ws + HB_OFF);
    unsigned* cnt  = (unsigned*)(ws + CNT_OFF);

    tau_pack<<<WP_HALFS / 256, 256, 0, stream>>>(W_h, U_x, b_h, b_x, Wp, bias);
    tau_init<<<(BATCH * NHID) / 256, 256, 0, stream>>>(hA, cnt);

    tau_scan<<<NBLOCKS, 256, 0, stream>>>(input, Wp, bias, hA, hB, cnt);

    // T_STEPS is even: final h is back in hA.
    tau_classifier<<<BATCH, 32, 0, stream>>>(hA, W_out, b_out, out);
}
